// QuantizationLayer_47536698032785
// MI455X (gfx1250) — matrix operand order repaired and checked by reading, not compiled, not run on hardware
//
#include <hip/hip_runtime.h>

#define CBINS 9
#define HDIM 80
#define HID 100
// padded MLP dims: K padded to 128 (4 k-tiles of 32; row K=100 carries b2),
// N padded to 112 (7 n-tiles of 16)
#define NT 7
#define KT 4

typedef __attribute__((ext_vector_type(16))) _Float16 v16h;
typedef __attribute__((ext_vector_type(8)))  float    v8f;
typedef __attribute__((ext_vector_type(2)))  float    v2f;

// A-fragment K mapping for 16-bit 16x32 A (ISA 7.12.2):
//   lanes 0-15 : VGPR0-3 -> K=0..7,   VGPR4-7 -> K=16..23
//   lanes 16-31: VGPR0-3 -> K=8..15,  VGPR4-7 -> K=24..31
__device__ __forceinline__ int a_kmap(int i, int hi) {
    int k = (i < 8) ? i : (i + 8);
    return k + (hi ? 8 : 0);
}

// Packed reduction merge: lanes with (lane&mask)==0 accumulate `a`'s row,
// lanes with (lane&mask)!=0 accumulate `b`'s row; exchange partner's value.
__device__ __forceinline__ float merge2(float a, float b, bool sel, int mask) {
    float x    = sel ? a : b;   // value handed to the partner lane
    float keep = sel ? b : a;   // value we keep accumulating
    return keep + __shfl_xor(x, mask, 32);
}

__global__ void zero_out_kernel(float* out, int n) {
    int i = blockIdx.x * blockDim.x + threadIdx.x;
    if (i < n) out[i] = 0.0f;
}

__global__ __launch_bounds__(256)
void mlp_voxel_kernel(const float* __restrict__ events,
                      const float* __restrict__ W1,
                      const float* __restrict__ b1,
                      const float* __restrict__ W2,
                      const float* __restrict__ b2,
                      const float* __restrict__ W3,
                      const float* __restrict__ b3,
                      float* __restrict__ out,
                      int N) {
    __shared__ __align__(32) _Float16 sB [NT][KT][32][16];  // 28672 B
    __shared__ __align__(32) _Float16 sW1[KT][32][16];      //  4096 B
    __shared__ __align__(32) _Float16 sb1[KT][32][16];      //  4096 B

    const int tid = threadIdx.x;

    // ---- one-time swizzle of W2 (+ b2 as row K=100) into B-fragment layout ----
    for (int idx = tid; idx < NT * KT * 32 * 16; idx += blockDim.x) {
        int i    = idx & 15;
        int lane = (idx >> 4) & 31;
        int kt   = (idx >> 9) & 3;
        int nt   = idx >> 11;
        int n = nt * 16 + (lane & 15);
        int k = kt * 32 + a_kmap(i, lane >= 16);
        float v = 0.0f;
        if (n < HID) {
            if (k < HID)       v = W2[k * HID + n];
            else if (k == HID) v = b2[n];           // bias row: A supplies 1.0 here
        }
        sB[nt][kt][lane][i] = (_Float16)v;
    }
    // ---- one-time swizzle of W1/b1 into A-fragment K pattern ----
    // slot K=100 gets (W1=0, b1=1) so lrelu(ts*0+1)=1 feeds the bias row.
    for (int idx = tid; idx < KT * 32 * 16; idx += blockDim.x) {
        int i    = idx & 15;
        int lane = (idx >> 4) & 31;
        int kt   = idx >> 9;
        int k = kt * 32 + a_kmap(i, lane >= 16);
        sW1[kt][lane][i] = (k < HID) ? (_Float16)W1[k] : (_Float16)0.0f;
        _Float16 bv = (_Float16)0.0f;
        if (k < HID)       bv = (_Float16)b1[k];
        else if (k == HID) bv = (_Float16)1.0f;
        sb1[kt][lane][i] = bv;
    }
    __syncthreads();

    const int lane = tid & 31;
    const int col  = lane & 15;   // event row M this lane owns (A layout: M = lane&15)

    // per-lane column constants (C/D layout: N = nt*16 + col)
    // acc already includes b2 (bias row), so: lrelu(acc)*w3 = acc*c1 + |acc|*c2,
    // c1 = 0.55*w3, c2 = 0.45*w3.
    float c1c[NT], c2c[NT];
#pragma unroll
    for (int nt = 0; nt < NT; ++nt) {
        int n = nt * 16 + col;
        float w3v = (n < HID) ? W3[n] : 0.0f;
        c1c[nt] = 0.55f * w3v;
        c2c[nt] = 0.45f * w3v;
    }
    const float b3v = b3[0];

    // packed reduction leaves row (lane&7 [+8 in high half]) in this lane;
    // lanes with bit3 set hold duplicates and do not emit.
    const int  Mself   = (lane & 7) + ((lane >= 16) ? 8 : 0);
    const bool emitter = (lane & 8) == 0;
    const bool selA = (lane & 1) != 0;
    const bool selB = (lane & 2) != 0;
    const bool selC = (lane & 4) != 0;

    // k-tile 3: only K=96..99 vary per bin; K>=100 slots are bin-invariant
    // (value = b1 slot itself: 1.0 at K=100, 0 elsewhere; lrelu is identity there).
    const v16h w1f3 = *(const v16h*)&sW1[3][lane][0];
    const v16h b1f3 = *(const v16h*)&sb1[3][lane][0];
    v16h af3 = b1f3;   // elements 0..3 are overwritten every bin

    const int wavesPerBlock = blockDim.x >> 5;
    const long gwave  = (long)blockIdx.x * wavesPerBlock + (tid >> 5);
    const long nwaves = (long)gridDim.x * wavesPerBlock;

    for (long base = gwave * 16; base < (long)N; base += nwaves * 16) {
        // N is a multiple of 16, so all 16 rows are valid.
        const float4 ev = ((const float4*)events)[base + col];
        const float xx = ev.x, tt = ev.y, pp = ev.z, bb = ev.w;
        const int obase = (int)bb * (2 * CBINS * HDIM) + (int)pp * HDIM + (int)xx - 1;
        const float coef = 0.01f * tt;   // fold t and the final /100

        // fetch the emitted row's event data once per 16-event group
        const float c_self  = __shfl(coef, Mself, 32);
        const int   o_self  = __shfl(obase, Mself, 32);
        const float cb_self = c_self * b3v;

#pragma unroll 1
        for (int ib = 0; ib < CBINS; ++ib) {
            const float ts = tt - (float)ib * (1.0f / (CBINS - 1));
            const _Float16 tsh = (_Float16)ts;

            // layer 1 -> A fragments (f16), lrelu(z) = max(z, 0.1z)
            v16h af[3];
#pragma unroll
            for (int kt = 0; kt < 3; ++kt) {
                v16h w1f = *(const v16h*)&sW1[kt][lane][0];
                v16h b1f = *(const v16h*)&sb1[kt][lane][0];
                v16h z;
#pragma unroll
                for (int j = 0; j < 16; ++j) {
                    _Float16 zz = tsh * w1f[j] + b1f[j];
                    z[j] = __builtin_fmaxf16(zz, zz * (_Float16)0.1f);
                }
                af[kt] = z;
            }
            // k-tile 3: only the 4 live slots
#pragma unroll
            for (int j = 0; j < 4; ++j) {
                _Float16 zz = tsh * w1f3[j] + b1f3[j];
                af3[j] = __builtin_fmaxf16(zz, zz * (_Float16)0.1f);
            }

            // layer 2 (+bias row): 7 n-tiles x 4 k-tiles of v_wmma_f32_16x16x32_f16
            v8f acc[NT] = {};
#pragma unroll
            for (int nt = 0; nt < NT; ++nt) {
#pragma unroll
                for (int kt = 0; kt < KT; ++kt) {
                    v16h bf = *(const v16h*)&sB[nt][kt][lane][0];
                    v16h a  = (kt < 3) ? af[kt] : af3;
                    acc[nt] = __builtin_amdgcn_wmma_f32_16x16x32_f16(
                        false, a, false, bf, (short)0, acc[nt], false, false);
                }
            }

            // epilogue: pr += acc*c1 + |acc|*c2   (acc already includes b2)
            v2f pr2[4] = {};
#pragma unroll
            for (int nt = 0; nt < NT; ++nt) {
                v2f c1v; c1v.x = c1c[nt]; c1v.y = c1c[nt];
                const float c2s = c2c[nt];
#pragma unroll
                for (int q = 0; q < 4; ++q) {
                    v2f a; a.x = acc[nt][2 * q]; a.y = acc[nt][2 * q + 1];
                    pr2[q] = a * c1v + pr2[q];    // v_pk_fma_f32
                    pr2[q].x = __builtin_fmaf(__builtin_fabsf(a.x), c2s, pr2[q].x);
                    pr2[q].y = __builtin_fmaf(__builtin_fabsf(a.y), c2s, pr2[q].y);
                }
            }

            // packed butterfly: 8 row-sums over 16 lanes; row (lane&7) ends here.
            float s01 = merge2(pr2[0].x, pr2[0].y, selA, 1);
            float s23 = merge2(pr2[1].x, pr2[1].y, selA, 1);
            float s45 = merge2(pr2[2].x, pr2[2].y, selA, 1);
            float s67 = merge2(pr2[3].x, pr2[3].y, selA, 1);
            float t03 = merge2(s01, s23, selB, 2);
            float t47 = merge2(s45, s67, selB, 2);
            float u07 = merge2(t03, t47, selC, 4);
            float v   = u07 + __shfl_xor(u07, 8, 32);

            if (emitter) {
                atomicAdd(out + o_self + ib * (2 * HDIM),
                          __builtin_fmaf(v, c_self, cb_self));
            }
        }
    }
}

extern "C" void kernel_launch(void* const* d_in, const int* in_sizes, int n_in,
                              void* d_out, int out_size, void* d_ws, size_t ws_size,
                              hipStream_t stream) {
    (void)n_in; (void)d_ws; (void)ws_size;
    const float* events = (const float*)d_in[0];
    const float* W1 = (const float*)d_in[1];
    const float* b1 = (const float*)d_in[2];
    const float* W2 = (const float*)d_in[3];
    const float* b2 = (const float*)d_in[4];
    const float* W3 = (const float*)d_in[5];
    const float* b3 = (const float*)d_in[6];
    float* out = (float*)d_out;
    const int N = in_sizes[0] / 4;

    // output is accumulated with atomics -> must start from zero every call
    zero_out_kernel<<<(out_size + 255) / 256, 256, 0, stream>>>(out, out_size);

    const long waveIters = (N + 15) / 16;
    long blocks = (waveIters + 7) / 8;          // 8 waves per 256-thread block
    if (blocks > 2048) blocks = 2048;
    if (blocks < 1) blocks = 1;
    mlp_voxel_kernel<<<(int)blocks, 256, 0, stream>>>(events, W1, b1, W2, b2, W3, b3, out, N);
}
